// MYMPNN_79276506349844
// MI455X (gfx1250) — matrix operand order repaired and checked by reading, not compiled, not run on hardware
//
#include <hip/hip_runtime.h>
#include <math.h>

typedef __attribute__((ext_vector_type(16))) _Float16 v16h;
typedef __attribute__((ext_vector_type(8)))  float    v8f;

#define B_    16
#define N_    48
#define HID_  64
#define TGT_  128
#define NE_   (B_*N_*N_)   /* 36864 edges */
#define NN_   (B_*N_)      /* 768 nodes  */

// ---------- WMMA fragment helpers (gfx1250 wave32, 16x16x32 f16) ----------
__device__ __forceinline__ int kbaseA(int v, int hf) {
  return (v < 4 ? 2*v : 16 + 2*(v-4)) + 8*hf;
}
__device__ __forceinline__ int kbaseB(int v, int hf) {
  return (v < 4 ? 2*v : 16 + 2*(v-4)) + 8*hf;
}

__device__ __forceinline__ v16h load_a(const _Float16* rowp, int k0, int hf) {
  v16h a;
#pragma unroll
  for (int v = 0; v < 8; ++v) {
    int k = k0 + kbaseA(v, hf);
    a[2*v]   = rowp[k];
    a[2*v+1] = rowp[k+1];
  }
  return a;
}

// One 16x16 output tile: A from LDS (row-major f16, leading dim ldm),
// B from pre-swizzled global fragments (tile-major: (ks*NT+nt)*32 + lane).
__device__ __forceinline__ v8f gemm_tile(const _Float16* Alds, int ldm, int mtile,
                                         const v16h* fragB, int NT, int nt,
                                         int ksteps, int lane) {
  int hf = lane >> 4;
  const _Float16* rowp = Alds + (mtile*16 + (lane & 15)) * ldm;
  v8f acc = {};
  for (int ks = 0; ks < ksteps; ++ks) {
    v16h a = load_a(rowp, ks*32, hf);
    v16h b = fragB[(ks*NT + nt)*32 + lane];
    acc = __builtin_amdgcn_wmma_f32_16x16x32_f16(false, a, false, b,
                                                 (short)0, acc, false, false);
  }
  return acc;
}

__device__ __forceinline__ void store_relu_f16(v8f acc, const float* bias,
                                               _Float16* Y, int ldm,
                                               int mtile, int nt, int lane) {
  int hf = lane >> 4, col = nt*16 + (lane & 15);
  float bv = bias[col];
#pragma unroll
  for (int j = 0; j < 8; ++j) {
    float x = acc[j] + bv;
    Y[(mtile*16 + j + 8*hf)*ldm + col] = (_Float16)fmaxf(x, 0.f);
  }
}

// ---------- prep: prefix sums of msg_w2 over 16-element blocks ----------
// Cmat[k, jd*64+m] = sum_{t < 16*(jd+1)} msg_w2[k, m*64+t]   (row-major 128x256)
// bC[jd*64+m]      = sum_{t < 16*(jd+1)} msg_b2[m*64+t]
__global__ __launch_bounds__(256) void prep_cmat_k(const float* __restrict__ W2,
                                                   const float* __restrict__ b2,
                                                   float* __restrict__ Cmat,
                                                   float* __restrict__ bC) {
  int idx = blockIdx.x*256 + threadIdx.x;
  if (idx >= 128*64) return;
  int k = idx >> 6, m = idx & 63;
  const float* src = W2 + (size_t)k*4096 + m*64;
  float s = 0.f;
  for (int jd = 0; jd < 4; ++jd) {
    for (int t = 0; t < 16; ++t) s += src[jd*16 + t];
    Cmat[k*256 + jd*64 + m] = s;
  }
  if (k == 0) {
    const float* bs = b2 + m*64;
    float sb = 0.f;
    for (int jd = 0; jd < 4; ++jd) {
      for (int t = 0; t < 16; ++t) sb += bs[jd*16 + t];
      bC[jd*64 + m] = sb;
    }
  }
}

// ---------- prep: convert f32 [KxN] weight to f16 B-fragment order ----------
__global__ __launch_bounds__(32) void prep_frag_k(const float* __restrict__ src,
                                                  int K, int N,
                                                  _Float16* __restrict__ dst) {
  int NT = N >> 4;
  int nt = blockIdx.x % NT, kt = blockIdx.x / NT;
  int lane = threadIdx.x, hf = lane >> 4, n = nt*16 + (lane & 15);
  _Float16* out = dst + ((size_t)blockIdx.x*32 + lane)*16;
#pragma unroll
  for (int v = 0; v < 8; ++v) {
    int k = kt*32 + kbaseB(v, hf);
    out[2*v]   = (k   < K) ? (_Float16)src[(size_t)k*N + n]     : (_Float16)0.f;
    out[2*v+1] = (k+1 < K) ? (_Float16)src[(size_t)(k+1)*N + n] : (_Float16)0.f;
  }
}

// ---------- init: h_first (padded h0), mask ----------
__global__ __launch_bounds__(64) void init_h_k(const float* __restrict__ h0,
                                               float* __restrict__ hfirst,
                                               float* __restrict__ hb0,
                                               float* __restrict__ mask) {
  int node = blockIdx.x, c = threadIdx.x;
  float v = (c < 32) ? h0[node*32 + c] : 0.f;
  hfirst[node*64 + c] = v;
  hb0[node*64 + c]    = v;
  __shared__ float s[64];
  s[c] = v;
  __syncthreads();
  if (c == 0) {
    float sum = 0.f;
    for (int t = 0; t < 32; ++t) sum += s[t];
    mask[node] = (sum > 0.f) ? 1.f : 0.f;
  }
}

// ---------- stage1: per-edge MLP folded into Q = X2 @ Cmat (once) ----------
// Q[e, jd*64+m] = x2[e] . Cmat[:, jd*64+m];  64 edge rows per block.
__global__ __launch_bounds__(256) void stage1_k(const float* __restrict__ e,
                                                const v16h* __restrict__ fW0,
                                                const float* __restrict__ b0,
                                                const v16h* __restrict__ fW1,
                                                const float* __restrict__ b1,
                                                const v16h* __restrict__ fC,
                                                float* __restrict__ Q) {
  __shared__ _Float16 sE[64*32];
  __shared__ _Float16 sX1[64*128];
  __shared__ _Float16 sX2[64*128];
  int tid = threadIdx.x, lane = tid & 31, w = tid >> 5, hf = lane >> 4;
  size_t e0 = (size_t)blockIdx.x * 64;
  for (int idx = tid; idx < 64*32; idx += 256) {
    int r = idx >> 5, k = idx & 31;
    sE[idx] = (_Float16)((k < 16) ? e[(e0 + r)*16 + k] : 0.f);
  }
  __syncthreads();
  int mt = w >> 1;
  // X1 = relu(E @ W0 + b0), K=16 (zero-padded to 32)
  for (int t = 0; t < 4; ++t) {
    int nt = (w & 1)*4 + t;
    v8f acc = gemm_tile(sE, 32, mt, fW0, 8, nt, 1, lane);
    store_relu_f16(acc, b0, sX1, 128, mt, nt, lane);
  }
  __syncthreads();
  // X2 = relu(X1 @ W1 + b1), K=128
  for (int t = 0; t < 4; ++t) {
    int nt = (w & 1)*4 + t;
    v8f acc = gemm_tile(sX1, 128, mt, fW1, 8, nt, 4, lane);
    store_relu_f16(acc, b1, sX2, 128, mt, nt, lane);
  }
  __syncthreads();
  // Q = X2 @ Cmat, N=256
  for (int t = 0; t < 8; ++t) {
    int nt = (w & 1)*8 + t;
    v8f acc = gemm_tile(sX2, 128, mt, fC, 16, nt, 4, lane);
    int col = nt*16 + (lane & 15);
#pragma unroll
    for (int j = 0; j < 8; ++j)
      Q[(e0 + mt*16 + j + 8*hf)*256 + col] = acc[j];
  }
}

// ---------- per-layer: message combine + masked-sum + GRU ----------
// block = one node (b,q), 64 threads = hidden channel m
__global__ __launch_bounds__(64) void layer_k(const float* __restrict__ g,
                                              const float* __restrict__ Q,
                                              const float* __restrict__ bC,
                                              const float* __restrict__ hin,
                                              float* __restrict__ hout,
                                              const float* __restrict__ mask,
                                              const float* __restrict__ wih,
                                              const float* __restrict__ whh,
                                              const float* __restrict__ bih,
                                              const float* __restrict__ bhh) {
  int node = blockIdx.x;
  int b = node / N_, q = node - b*N_;
  int m = threadIdx.x;
  int jd = 2 - (q % 3);          // prefix index: L0 in {48,32,16} -> jd {2,1,0}
  int c0 = (4*q) / 3;            // first hidden channel of the segment pair
  float bc0 = bC[jd*64 + m];
  float bc1 = bC[192 + m] - bc0;
  __shared__ float sAgg[64];
  __shared__ float sH[64];
  float agg = 0.f;
  const float* grow = g + (size_t)b*N_*N_ + q;
  for (int i = 0; i < N_; ++i) {
    float gv = grow[i*N_];       // uniform across block
    if (gv != 0.f) {
      size_t erow = ((size_t)(b*N_ + i)*N_ + q)*256;
      float q0 = Q[erow + jd*64 + m];
      float p0 = q0 + bc0;
      float p1 = (Q[erow + 192 + m] - q0) + bc1;
      float hA = hin[(b*N_ + i)*64 + c0];
      float hB = hin[(b*N_ + i)*64 + c0 + 1];
      agg += gv * (hA*p0 + hB*p1);
    }
  }
  float hm = hin[node*64 + m];
  sAgg[m] = agg;
  sH[m]   = hm;
  __syncthreads();
  float gir = bih[m], giz = bih[64 + m], gin = bih[128 + m];
  float ghr = bhh[m], ghz = bhh[64 + m], ghn = bhh[128 + m];
  for (int k = 0; k < 64; ++k) {
    float x = sAgg[k], h = sH[k];
    gir += x * wih[m*64 + k];
    giz += x * wih[(64 + m)*64 + k];
    gin += x * wih[(128 + m)*64 + k];
    ghr += h * whh[m*64 + k];
    ghz += h * whh[(64 + m)*64 + k];
    ghn += h * whh[(128 + m)*64 + k];
  }
  float r = 1.f / (1.f + __expf(-(gir + ghr)));
  float z = 1.f / (1.f + __expf(-(giz + ghz)));
  float n = tanhf(gin + r*ghn);
  float hnew = (1.f - z)*n + z*hm;
  hout[node*64 + m] = mask[node] * hnew;
}

// ---------- readout: Net0(cat[h0,hT]) * Net1(hT), masked node sum ----------
__global__ __launch_bounds__(256) void readout_k(const float* __restrict__ hfirst,
                                                 const float* __restrict__ hT,
                                                 const float* __restrict__ mask,
    const v16h* fW00, const float* b00, const v16h* fW01, const float* b01,
    const v16h* fW02, const float* b02,
    const v16h* fW10, const float* b10, const v16h* fW11, const float* b11,
    const v16h* fW12, const float* b12, float* __restrict__ accum) {
  __shared__ _Float16 sA[64*128];
  __shared__ _Float16 sB[64*128];
  __shared__ float    sO0[64*128];
  int tid = threadIdx.x, lane = tid & 31, w = tid >> 5, hf = lane >> 4;
  int n0 = blockIdx.x * 64;
  for (int idx = tid; idx < 64*128; idx += 256) {
    int r = idx >> 7, c = idx & 127;
    float v = (c < 64) ? hfirst[(n0 + r)*64 + c] : hT[(n0 + r)*64 + (c - 64)];
    sA[idx] = (_Float16)v;
  }
  __syncthreads();
  int mt = w >> 1;
  for (int t = 0; t < 4; ++t) { int nt = (w & 1)*4 + t;
    v8f acc = gemm_tile(sA, 128, mt, fW00, 8, nt, 4, lane);
    store_relu_f16(acc, b00, sB, 128, mt, nt, lane); }
  __syncthreads();
  for (int t = 0; t < 4; ++t) { int nt = (w & 1)*4 + t;
    v8f acc = gemm_tile(sB, 128, mt, fW01, 8, nt, 4, lane);
    store_relu_f16(acc, b01, sA, 128, mt, nt, lane); }
  __syncthreads();
  for (int t = 0; t < 4; ++t) { int nt = (w & 1)*4 + t;
    v8f acc = gemm_tile(sA, 128, mt, fW02, 8, nt, 4, lane);
    int col = nt*16 + (lane & 15);
#pragma unroll
    for (int j = 0; j < 8; ++j)
      sO0[(mt*16 + j + 8*hf)*128 + col] = acc[j] + b02[col];
  }
  __syncthreads();
  // path 2: Net1(hT), K=64
  for (int idx = tid; idx < 64*64; idx += 256) {
    int r = idx >> 6, c = idx & 63;
    sB[r*64 + c] = (_Float16)hT[(n0 + r)*64 + c];
  }
  __syncthreads();
  for (int t = 0; t < 4; ++t) { int nt = (w & 1)*4 + t;
    v8f acc = gemm_tile(sB, 64, mt, fW10, 8, nt, 2, lane);
    store_relu_f16(acc, b10, sA, 128, mt, nt, lane); }
  __syncthreads();
  for (int t = 0; t < 4; ++t) { int nt = (w & 1)*4 + t;
    v8f acc = gemm_tile(sA, 128, mt, fW11, 8, nt, 4, lane);
    store_relu_f16(acc, b11, sB, 128, mt, nt, lane); }
  __syncthreads();
  for (int t = 0; t < 4; ++t) { int nt = (w & 1)*4 + t;
    v8f acc = gemm_tile(sB, 128, mt, fW12, 8, nt, 4, lane);
    int col = nt*16 + (lane & 15);
#pragma unroll
    for (int j = 0; j < 8; ++j) {
      int row = mt*16 + j + 8*hf;
      int node = n0 + row;
      float val = mask[node] * sO0[row*128 + col] * (acc[j] + b12[col]);
      atomicAdd(&accum[(node / N_)*TGT_ + col], val);
    }
  }
}

__global__ void zero_k(float* p, int n) {
  int i = blockIdx.x*blockDim.x + threadIdx.x;
  if (i < n) p[i] = 0.f;
}
__global__ void final_k(const float* __restrict__ accum, float* __restrict__ out, int n) {
  int i = blockIdx.x*blockDim.x + threadIdx.x;
  if (i < n) out[i] = 1.f / (1.f + expf(-accum[i]));
}

extern "C" void kernel_launch(void* const* d_in, const int* in_sizes, int n_in,
                              void* d_out, int out_size, void* d_ws, size_t ws_size,
                              hipStream_t stream) {
  const float* g      = (const float*)d_in[0];
  const float* h0     = (const float*)d_in[1];
  const float* e      = (const float*)d_in[2];
  const float* msg_w0 = (const float*)d_in[3];
  const float* msg_b0 = (const float*)d_in[4];
  const float* msg_w1 = (const float*)d_in[5];
  const float* msg_b1 = (const float*)d_in[6];
  const float* msg_w2 = (const float*)d_in[7];
  const float* msg_b2 = (const float*)d_in[8];
  const float* gwih = (const float*)d_in[9];
  const float* gwhh = (const float*)d_in[10];
  const float* gbih = (const float*)d_in[11];
  const float* gbhh = (const float*)d_in[12];
  const float* r0w0 = (const float*)d_in[13]; const float* r0b0 = (const float*)d_in[14];
  const float* r0w1 = (const float*)d_in[15]; const float* r0b1 = (const float*)d_in[16];
  const float* r0w2 = (const float*)d_in[17]; const float* r0b2 = (const float*)d_in[18];
  const float* r1w0 = (const float*)d_in[19]; const float* r1b0 = (const float*)d_in[20];
  const float* r1w1 = (const float*)d_in[21]; const float* r1b1 = (const float*)d_in[22];
  const float* r1w2 = (const float*)d_in[23]; const float* r1b2 = (const float*)d_in[24];

  char* ws = (char*)d_ws;
  size_t off = 0;
  auto take = [&](size_t bytes) -> char* {
    char* p = ws + off;
    off = (off + bytes + 255) & ~(size_t)255;
    return p;
  };
  float* Q      = (float*)take((size_t)NE_*256*4);   // 37.75 MB
  float* accum  = (float*)take((size_t)B_*TGT_*4);
  float* hfirst = (float*)take((size_t)NN_*64*4);
  float* hb0    = (float*)take((size_t)NN_*64*4);
  float* hb1    = (float*)take((size_t)NN_*64*4);
  float* mask   = (float*)take((size_t)NN_*4);
  float* Cmat   = (float*)take((size_t)128*256*4);
  float* bC     = (float*)take((size_t)256*4);
  _Float16* fW0  = (_Float16*)take((size_t)1*8*32*16*2);
  _Float16* fW1  = (_Float16*)take((size_t)4*8*32*16*2);
  _Float16* fC   = (_Float16*)take((size_t)4*16*32*16*2);
  _Float16* fR00 = (_Float16*)take((size_t)4*8*32*16*2);
  _Float16* fR01 = (_Float16*)take((size_t)4*8*32*16*2);
  _Float16* fR02 = (_Float16*)take((size_t)4*8*32*16*2);
  _Float16* fR10 = (_Float16*)take((size_t)2*8*32*16*2);
  _Float16* fR11 = (_Float16*)take((size_t)4*8*32*16*2);
  _Float16* fR12 = (_Float16*)take((size_t)4*8*32*16*2);

  zero_k<<<(B_*TGT_ + 255)/256, 256, 0, stream>>>(accum, B_*TGT_);
  prep_cmat_k<<<(128*64 + 255)/256, 256, 0, stream>>>(msg_w2, msg_b2, Cmat, bC);
  prep_frag_k<<<1*8,  32, 0, stream>>>(msg_w0, 16, 128, fW0);
  prep_frag_k<<<4*8,  32, 0, stream>>>(msg_w1, 128, 128, fW1);
  prep_frag_k<<<4*16, 32, 0, stream>>>(Cmat,   128, 256, fC);
  prep_frag_k<<<4*8,  32, 0, stream>>>(r0w0,   128, 128, fR00);
  prep_frag_k<<<4*8,  32, 0, stream>>>(r0w1,   128, 128, fR01);
  prep_frag_k<<<4*8,  32, 0, stream>>>(r0w2,   128, 128, fR02);
  prep_frag_k<<<2*8,  32, 0, stream>>>(r1w0,    64, 128, fR10);
  prep_frag_k<<<4*8,  32, 0, stream>>>(r1w1,   128, 128, fR11);
  prep_frag_k<<<4*8,  32, 0, stream>>>(r1w2,   128, 128, fR12);
  init_h_k<<<NN_, 64, 0, stream>>>(h0, hfirst, hb0, mask);
  stage1_k<<<NE_/64, 256, 0, stream>>>(e, (const v16h*)fW0, msg_b0,
                                       (const v16h*)fW1, msg_b1,
                                       (const v16h*)fC, Q);
  layer_k<<<NN_, 64, 0, stream>>>(g, Q, bC, hb0, hb1, mask, gwih, gwhh, gbih, gbhh);
  layer_k<<<NN_, 64, 0, stream>>>(g, Q, bC, hb1, hb0, mask, gwih, gwhh, gbih, gbhh);
  layer_k<<<NN_, 64, 0, stream>>>(g, Q, bC, hb0, hb1, mask, gwih, gwhh, gbih, gbhh);
  readout_k<<<NN_/64, 256, 0, stream>>>(hfirst, hb1, mask,
      (const v16h*)fR00, r0b0, (const v16h*)fR01, r0b1, (const v16h*)fR02, r0b2,
      (const v16h*)fR10, r1b0, (const v16h*)fR11, r1b1, (const v16h*)fR12, r1b2,
      accum);
  final_k<<<(B_*TGT_ + 255)/256, 256, 0, stream>>>(accum, (float*)d_out, B_*TGT_);
}
